// BigBirdRegressor_MLP_Reduced_42150809043593
// MI455X (gfx1250) — matrix operand order repaired and checked by reading, not compiled, not run on hardware
//
#include <hip/hip_runtime.h>
#include <math.h>

typedef __attribute__((ext_vector_type(2)))  float    v2f;
typedef __attribute__((ext_vector_type(4)))  float    v4f;
typedef __attribute__((ext_vector_type(8)))  float    v8f;
typedef __attribute__((ext_vector_type(16))) _Float16 v16h;

#define B_    4
#define S_    16384
#define H_    3
#define NH    3
#define NB    256
#define BLK   64
#define NM    254          // NB - 2
#define FF    12
#define H1DIM 1000
#define H2DIM 500
#define NTOK  (B_*S_)

// ---- workspace layout (float offsets) ----
#define OFF_X    0u
#define OFF_Q    196608u
#define OFF_K    393216u
#define OFF_V    589824u
#define OFF_CTX  786432u
#define OFF_RED  983040u          // 4*16384
#define OFF_H1   1048576u         // 4*1000 (atomic accum)
#define OFF_A16  1052672u         // 16*1024 padded fc2 input
#define OFF_H2   1069056u         // 4*500
#define OFF_H2N  1071104u         // 4*500
#define OFF_KIDX 1073152u         // nH*nM*8 ints

__device__ __forceinline__ void ln3(const float* in, const float* gb, float* out) {
    float mu = (in[0] + in[1] + in[2]) * (1.0f / 3.0f);
    float d0 = in[0] - mu, d1 = in[1] - mu, d2 = in[2] - mu;
    float var = (d0 * d0 + d1 * d1 + d2 * d2) * (1.0f / 3.0f);
    float r = rsqrtf(var + 1e-12f);
    out[0] = d0 * r * gb[0] + gb[3];
    out[1] = d1 * r * gb[1] + gb[4];
    out[2] = d2 * r * gb[2] + gb[5];
}

__device__ __forceinline__ float gelu_tanh(float u) {
    return 0.5f * u * (1.0f + tanhf(0.7978845608028654f * (u + 0.044715f * u * u * u)));
}

// ---------------- embedding layernorm ----------------
__global__ void k_embed(const float* __restrict__ inp, const float* __restrict__ pos,
                        const float* __restrict__ typ, const float* __restrict__ lngb,
                        float* __restrict__ X) {
    int t = blockIdx.x * blockDim.x + threadIdx.x;
    if (t >= NTOK) return;
    int s = t % S_;
    float v[3], o[3];
    for (int c = 0; c < 3; ++c) v[c] = inp[t * 3 + c] + pos[s * 3 + c] + typ[c];
    ln3(v, lngb, o);
    for (int c = 0; c < 3; ++c) X[t * 3 + c] = o[c];
}

// ---------------- per-token q,k,v (dh == 1) ----------------
__global__ void k_qkv(const float* __restrict__ X, const float* __restrict__ qkvw,
                      const float* __restrict__ qkvb, float* __restrict__ Q,
                      float* __restrict__ K, float* __restrict__ V, int l) {
    int t = blockIdx.x * blockDim.x + threadIdx.x;
    if (t >= NTOK) return;
    int b = t / S_, s = t % S_;
    float x0 = X[t * 3], x1 = X[t * 3 + 1], x2 = X[t * 3 + 2];
    const float* wq = qkvw + (l * 3 + 0) * 9;
    const float* wk = qkvw + (l * 3 + 1) * 9;
    const float* wv = qkvw + (l * 3 + 2) * 9;
    const float* bq = qkvb + (l * 3 + 0) * 3;
    const float* bk = qkvb + (l * 3 + 1) * 3;
    const float* bv = qkvb + (l * 3 + 2) * 3;
    for (int h = 0; h < 3; ++h) {
        int o = (b * NH + h) * S_ + s;
        Q[o] = x0 * wq[h] + x1 * wq[3 + h] + x2 * wq[6 + h] + bq[h];
        K[o] = x0 * wk[h] + x1 * wk[3 + h] + x2 * wk[6 + h] + bk[h];
        V[o] = x0 * wv[h] + x1 * wv[3 + h] + x2 * wv[6 + h] + bv[h];
    }
}

// ---------------- sparse key-block index table ----------------
__global__ void k_keyidx(int* __restrict__ kidx) {
    int t = blockIdx.x * blockDim.x + threadIdx.x;
    if (t >= NH * NM) return;
    int h = t / NM, m = t % NM;
    int* o = kidx + t * 8;
    o[0] = 0; o[1] = m; o[2] = m + 1; o[3] = m + 2; o[4] = NB - 1;
    unsigned st = 0x9E3779B9u ^ (unsigned)(h * 0x85EBCA6B) ^ (unsigned)(m * 0xC2B2AE35);
    int p0 = -1, p1 = -1;
    for (int j = 0; j < 3; ++j) {
        int c;
        for (;;) {
            st = st * 1664525u + 1013904223u;
            c = (int)((st >> 10) & 255u);
            if (c == 0 || c == m || c == m + 1 || c == m + 2 || c == NB - 1) continue;
            if (c == p0 || c == p1) continue;
            break;
        }
        if (j == 0) p0 = c; else if (j == 1) p1 = c;
        o[5 + j] = c;
    }
}

// ---------------- middle-block sparse attention ----------------
__global__ void k_attn_mid(const float* __restrict__ Q, const float* __restrict__ K,
                           const float* __restrict__ V, const int* __restrict__ kidx,
                           float* __restrict__ CTX) {
    __shared__ float sk[512], sv[512];
    int wg = blockIdx.x;                  // b*NH*NM blocks
    int m = wg % NM, h = (wg / NM) % NH, b = wg / (NM * NH);
    const int* idx = kidx + (h * NM + m) * 8;
    int base = (b * NH + h) * S_;
    int tid = threadIdx.x;                // 64 threads
    for (int j = 0; j < 8; ++j) {
        int kb = idx[j];
        sk[j * 64 + tid] = K[base + kb * 64 + tid];
        sv[j * 64 + tid] = V[base + kb * 64 + tid];
    }
    __syncthreads();
    float a = Q[base + (m + 1) * 64 + tid];   // scale = 1/sqrt(1) = 1
    float mmax = -INFINITY, l = 0.f, acc = 0.f;
    for (int k = 0; k < 512; ++k) {
        float s = a * sk[k];
        float nm = fmaxf(mmax, s);
        float corr = __expf(mmax - nm);
        float p = __expf(s - nm);
        l = l * corr + p;
        acc = acc * corr + p * sv[k];
        mmax = nm;
    }
    CTX[base + (m + 1) * 64 + tid] = acc / l;
}

// ---------------- global-block attention (split-K flash) ----------------
#define GCH 8
__global__ void k_attn_glob(const float* __restrict__ Q, const float* __restrict__ K,
                            const float* __restrict__ V, float* __restrict__ CTX) {
    __shared__ float sm[GCH * 64], sl[GCH * 64], sa[GCH * 64];
    int wg = blockIdx.x;                  // B*NH*2 blocks
    int g = wg % 2, h = (wg / 2) % NH, b = wg / (2 * NH);
    int qb = g ? (NB - 1) : 0;
    int base = (b * NH + h) * S_;
    int tid = threadIdx.x;                // 512 threads
    int qi = tid & 63, ch = tid >> 6;
    float a = Q[base + qb * 64 + qi];
    int k0 = ch * (S_ / GCH), k1 = k0 + S_ / GCH;
    float mmax = -INFINITY, l = 0.f, acc = 0.f;
    for (int k = k0; k < k1; ++k) {
        float s = a * K[base + k];
        float nm = fmaxf(mmax, s);
        float corr = __expf(mmax - nm);
        float p = __expf(s - nm);
        l = l * corr + p;
        acc = acc * corr + p * V[base + k];
        mmax = nm;
    }
    sm[ch * 64 + qi] = mmax; sl[ch * 64 + qi] = l; sa[ch * 64 + qi] = acc;
    __syncthreads();
    if (tid < 64) {
        float M = -INFINITY;
        for (int c = 0; c < GCH; ++c) M = fmaxf(M, sm[c * 64 + tid]);
        float L = 0.f, A = 0.f;
        for (int c = 0; c < GCH; ++c) {
            float w = __expf(sm[c * 64 + tid] - M);
            L += w * sl[c * 64 + tid];
            A += w * sa[c * 64 + tid];
        }
        CTX[base + qb * 64 + tid] = A / L;
    }
}

// ---------------- attn-out + LN + FFN + LN (fused, per token) ----------------
__global__ void k_post(float* __restrict__ X, const float* __restrict__ CTX,
                       const float* __restrict__ attnw, const float* __restrict__ attnb,
                       const float* __restrict__ ln1, const float* __restrict__ fw1,
                       const float* __restrict__ fb1, const float* __restrict__ fw2,
                       const float* __restrict__ fb2, const float* __restrict__ ln2, int l) {
    int t = blockIdx.x * blockDim.x + threadIdx.x;
    if (t >= NTOK) return;
    int b = t / S_, s = t % S_;
    float ctx[3];
    for (int h = 0; h < 3; ++h) ctx[h] = CTX[(b * NH + h) * S_ + s];
    const float* aw = attnw + l * 9;
    const float* ab = attnb + l * 3;
    float xin[3], o[3], y[3];
    for (int c = 0; c < 3; ++c) {
        o[c] = ab[c];
        for (int h = 0; h < 3; ++h) o[c] += ctx[h] * aw[h * 3 + c];
        xin[c] = X[t * 3 + c] + o[c];
    }
    ln3(xin, ln1 + l * 6, y);
    const float* w1 = fw1 + l * 3 * FF;
    const float* b1 = fb1 + l * FF;
    const float* w2 = fw2 + l * FF * 3;
    const float* b2 = fb2 + l * 3;
    float o2[3] = {b2[0], b2[1], b2[2]};
    for (int j = 0; j < FF; ++j) {
        float u = b1[j] + y[0] * w1[j] + y[1] * w1[FF + j] + y[2] * w1[2 * FF + j];
        float gj = gelu_tanh(u);
        for (int c = 0; c < 3; ++c) o2[c] += gj * w2[j * 3 + c];
    }
    float zin[3], z[3];
    for (int c = 0; c < 3; ++c) zin[c] = y[c] + o2[c];
    ln3(zin, ln2 + l * 6, z);
    for (int c = 0; c < 3; ++c) X[t * 3 + c] = z[c];
}

// ---------------- reduce head + emit embedding/reduced outputs ----------------
__global__ void k_reduce(const float* __restrict__ X, const float* __restrict__ rw,
                         const float* __restrict__ rb, float* __restrict__ RED,
                         float* __restrict__ out_red, float* __restrict__ out_emb) {
    int t = blockIdx.x * blockDim.x + threadIdx.x;
    if (t >= NTOK) return;
    float x0 = X[t * 3], x1 = X[t * 3 + 1], x2 = X[t * 3 + 2];
    out_emb[t * 3] = x0; out_emb[t * 3 + 1] = x1; out_emb[t * 3 + 2] = x2;
    float r = x0 * rw[0] + x1 * rw[1] + x2 * rw[2] + rb[0];
    RED[t] = r;
    out_red[t] = r;
}

__global__ void k_zero(float* __restrict__ p, int n) {
    int t = blockIdx.x * blockDim.x + threadIdx.x;
    if (t < n) p[t] = 0.f;
}

// ---------------- fc1: (16x16384)@(16384x1000), K split 64 ways ----------------
// All loads are unconditional with clamped addresses:
//  * A rows >= 4 re-read row (m&3); they only affect C rows 4..15, never stored.
//  * B cols >= 1000 re-read col 999; they only affect C cols the store-guard drops.
// This keeps EXEC all-ones and branch-free through the WMMA loop.
#define FC1_NT 63
#define FC1_KSPLIT 64
#define FC1_KC (S_ / FC1_KSPLIT)

#if __has_builtin(__builtin_amdgcn_wmma_f32_16x16x4_f32)
#define HAVE_WMMA_F32 1
#else
#define HAVE_WMMA_F32 0
#endif

__global__ void k_fc1(const float* __restrict__ RED, const float* __restrict__ W,
                      float* __restrict__ H1) {
    int nt = blockIdx.x % FC1_NT;
    int kc = blockIdx.x / FC1_NT;
    int n0 = nt * 16;
    int kb = kc * FC1_KC;
    int lane = threadIdx.x;
    int m = lane & 15;
    int hi = lane >> 4;
    int col = n0 + m;
    bool cv = col < H1DIM;
    int colc = cv ? col : (H1DIM - 1);
    const float* arow = RED + (m & 3) * S_;
    v8f c = {};
#if HAVE_WMMA_F32
    for (int k0 = kb; k0 < kb + FC1_KC; k0 += 4) {
        int ka = k0 + hi * 2;
        v2f a = *(const v2f*)(arow + ka);                 // global_load_b64
        v2f bb;
        bb.x = W[ka * H1DIM + colc];
        bb.y = W[(ka + 1) * H1DIM + colc];
        int kp = ka + 8; kp = kp < S_ ? kp : (S_ - 1);    // clamped stream prefetch
        __builtin_prefetch(&W[kp * H1DIM + colc], 0, 1);  // global_prefetch_b8
        c = __builtin_amdgcn_wmma_f32_16x16x4_f32(false, a, false, bb, (short)0, c,
                                                  false, false);
    }
#else
    for (int k0 = kb; k0 < kb + FC1_KC; k0 += 32) {
        v16h a, bb;
        v4f a0 = *(const v4f*)(arow + k0 + hi * 8);
        v4f a1 = *(const v4f*)(arow + k0 + hi * 8 + 4);
        v4f a2 = *(const v4f*)(arow + k0 + 16 + hi * 8);
        v4f a3 = *(const v4f*)(arow + k0 + 16 + hi * 8 + 4);
        for (int j = 0; j < 4; ++j) {
            a[j]      = (_Float16)a0[j];
            a[4 + j]  = (_Float16)a1[j];
            a[8 + j]  = (_Float16)a2[j];
            a[12 + j] = (_Float16)a3[j];
        }
        for (int j = 0; j < 16; ++j) {
            int kk = k0 + (j < 8 ? hi * 8 + j : 16 + hi * 8 + (j - 8));
            bb[j] = (_Float16)W[kk * H1DIM + colc];
        }
        c = __builtin_amdgcn_wmma_f32_16x16x32_f16(false, a, false, bb, (short)0, c,
                                                   false, false);
    }
#endif
    if (hi == 0 && cv) {
        for (int r = 0; r < 4; ++r) atomicAdd(&H1[r * H1DIM + col], c[r]);
    }
}

// ---------------- bn1 + relu, emit padded 16x1024 fc2 input ----------------
__global__ void k_bn1(const float* __restrict__ H1, const float* __restrict__ bias,
                      const float* __restrict__ bn, float* __restrict__ A16) {
    int j = blockIdx.x * blockDim.x + threadIdx.x;
    if (j >= 1024) return;
    if (j < H1DIM) {
        float v[4], mu = 0.f;
        for (int r = 0; r < 4; ++r) { v[r] = H1[r * H1DIM + j] + bias[j]; mu += v[r]; }
        mu *= 0.25f;
        float var = 0.f;
        for (int r = 0; r < 4; ++r) { float d = v[r] - mu; var += d * d; }
        var *= 0.25f;
        float rs = rsqrtf(var + 1e-5f);
        float g = bn[j], be = bn[H1DIM + j];
        for (int r = 0; r < 4; ++r)
            A16[r * 1024 + j] = fmaxf(0.f, (v[r] - mu) * rs * g + be);
    } else {
        for (int r = 0; r < 4; ++r) A16[r * 1024 + j] = 0.f;
    }
    for (int r = 4; r < 16; ++r) A16[r * 1024 + j] = 0.f;
}

// ---------------- fc2: (16x1024)@(1000x500) via f16 WMMA ----------------
// K-padding (1000..1023) needs no masking: padded A16 columns are zero, so the
// clamped B rows multiply by zero. Clamped col likewise only feeds dropped C cols.
__global__ void k_fc2(const float* __restrict__ A16, const float* __restrict__ W,
                      float* __restrict__ H2) {
    int n0 = blockIdx.x * 16;             // 32 tiles cover 512 >= 500
    int lane = threadIdx.x;
    int m = lane & 15;
    int hi = lane >> 4;
    int col = n0 + m;
    bool cv = col < H2DIM;
    int colc = cv ? col : (H2DIM - 1);
    const float* arow = A16 + m * 1024;
    v8f c = {};
    for (int k0 = 0; k0 < 1024; k0 += 32) {
        v16h a, bb;
        v4f a0 = *(const v4f*)(arow + k0 + hi * 8);       // global_load_b128 x4
        v4f a1 = *(const v4f*)(arow + k0 + hi * 8 + 4);
        v4f a2 = *(const v4f*)(arow + k0 + 16 + hi * 8);
        v4f a3 = *(const v4f*)(arow + k0 + 16 + hi * 8 + 4);
        for (int j = 0; j < 4; ++j) {
            a[j]      = (_Float16)a0[j];
            a[4 + j]  = (_Float16)a1[j];
            a[8 + j]  = (_Float16)a2[j];
            a[12 + j] = (_Float16)a3[j];
        }
        for (int j = 0; j < 16; ++j) {
            int kk = k0 + (j < 8 ? hi * 8 + j : 16 + hi * 8 + (j - 8));
            kk = kk < H1DIM ? kk : (H1DIM - 1);
            bb[j] = (_Float16)W[kk * H2DIM + colc];
        }
        c = __builtin_amdgcn_wmma_f32_16x16x32_f16(false, a, false, bb, (short)0, c,
                                                   false, false);
    }
    if (hi == 0 && cv) {
        for (int r = 0; r < 4; ++r) H2[r * H2DIM + col] = c[r];
    }
}

// ---------------- bn2 + relu ----------------
__global__ void k_bn2(const float* __restrict__ H2, const float* __restrict__ bias,
                      const float* __restrict__ bn, float* __restrict__ H2N) {
    int j = blockIdx.x * blockDim.x + threadIdx.x;
    if (j >= H2DIM) return;
    float v[4], mu = 0.f;
    for (int r = 0; r < 4; ++r) { v[r] = H2[r * H2DIM + j] + bias[j]; mu += v[r]; }
    mu *= 0.25f;
    float var = 0.f;
    for (int r = 0; r < 4; ++r) { float d = v[r] - mu; var += d * d; }
    var *= 0.25f;
    float rs = rsqrtf(var + 1e-5f);
    float g = bn[j], be = bn[H2DIM + j];
    for (int r = 0; r < 4; ++r)
        H2N[r * H2DIM + j] = fmaxf(0.f, (v[r] - mu) * rs * g + be);
}

// ---------------- fc3: (4x500)@(500x1) ----------------
__global__ void k_fc3(const float* __restrict__ H2N, const float* __restrict__ w,
                      const float* __restrict__ b, float* __restrict__ logits) {
    int tid = threadIdx.x;
    if (tid < 4) {
        float s = b[0];
        for (int j = 0; j < H2DIM; ++j) s += H2N[tid * H2DIM + j] * w[j];
        logits[tid] = s;
    }
}

extern "C" void kernel_launch(void* const* d_in, const int* in_sizes, int n_in,
                              void* d_out, int out_size, void* d_ws, size_t ws_size,
                              hipStream_t stream) {
    (void)in_sizes; (void)n_in; (void)out_size; (void)ws_size;
    const float* inp   = (const float*)d_in[0];
    const float* pos   = (const float*)d_in[1];
    const float* typ   = (const float*)d_in[2];
    const float* lnemb = (const float*)d_in[3];
    const float* qkvw  = (const float*)d_in[4];
    const float* qkvb  = (const float*)d_in[5];
    const float* attnw = (const float*)d_in[6];
    const float* attnb = (const float*)d_in[7];
    const float* ln1   = (const float*)d_in[8];
    const float* fw1   = (const float*)d_in[9];
    const float* fb1   = (const float*)d_in[10];
    const float* fw2   = (const float*)d_in[11];
    const float* fb2   = (const float*)d_in[12];
    const float* ln2   = (const float*)d_in[13];
    const float* redw  = (const float*)d_in[14];
    const float* redb  = (const float*)d_in[15];
    const float* fc1w  = (const float*)d_in[16];
    const float* fc1b  = (const float*)d_in[17];
    const float* bn1   = (const float*)d_in[18];
    const float* fc2w  = (const float*)d_in[19];
    const float* fc2b  = (const float*)d_in[20];
    const float* bn2   = (const float*)d_in[21];
    const float* fc3w  = (const float*)d_in[22];
    const float* fc3b  = (const float*)d_in[23];

    float* ws  = (float*)d_ws;
    float* X   = ws + OFF_X;
    float* Q   = ws + OFF_Q;
    float* K   = ws + OFF_K;
    float* V   = ws + OFF_V;
    float* CTX = ws + OFF_CTX;
    float* RED = ws + OFF_RED;
    float* H1  = ws + OFF_H1;
    float* A16 = ws + OFF_A16;
    float* H2  = ws + OFF_H2;
    float* H2N = ws + OFF_H2N;
    int*   KIDX = (int*)(ws + OFF_KIDX);

    float* out       = (float*)d_out;
    float* out_logit = out;              // 4
    float* out_red   = out + 4;          // 65536
    float* out_emb   = out + 4 + NTOK;   // 196608

    dim3 tok_grid((NTOK + 255) / 256), tok_blk(256);

    k_keyidx<<<(NH * NM + 127) / 128, 128, 0, stream>>>(KIDX);
    k_embed<<<tok_grid, tok_blk, 0, stream>>>(inp, pos, typ, lnemb, X);

    for (int l = 0; l < 2; ++l) {
        k_qkv<<<tok_grid, tok_blk, 0, stream>>>(X, qkvw, qkvb, Q, K, V, l);
        k_attn_mid<<<B_ * NH * NM, 64, 0, stream>>>(Q, K, V, KIDX, CTX);
        k_attn_glob<<<B_ * NH * 2, 512, 0, stream>>>(Q, K, V, CTX);
        k_post<<<tok_grid, tok_blk, 0, stream>>>(X, CTX, attnw, attnb, ln1, fw1, fb1,
                                                 fw2, fb2, ln2, l);
    }

    k_reduce<<<tok_grid, tok_blk, 0, stream>>>(X, redw, redb, RED, out_red, out_emb);
    k_zero<<<(4 * H1DIM + 255) / 256, 256, 0, stream>>>(H1, 4 * H1DIM);
    k_fc1<<<FC1_NT * FC1_KSPLIT, 32, 0, stream>>>(RED, fc1w, H1);
    k_bn1<<<4, 256, 0, stream>>>(H1, fc1b, bn1, A16);
    k_fc2<<<32, 32, 0, stream>>>(A16, fc2w, H2);
    k_bn2<<<2, 256, 0, stream>>>(H2, fc2b, bn2, H2N);
    k_fc3<<<1, 32, 0, stream>>>(H2N, fc3w, fc3b, out_logit);
}
